// DenseMapsSoftmaxNeighborFinder_89026082111828
// MI455X (gfx1250) — matrix operand order repaired and checked by reading, not compiled, not run on hardware
//
#include <hip/hip_runtime.h>
#include <hip/hip_bf16.h>
#include <math.h>

// ---------------------------------------------------------------------------
// DenseMapsSoftmaxNeighborFinder for MI455X (gfx1250, wave32, WMMA)
//
// dists[i] = || Y[i] - softmax_j(-||Y_i - X_j||^2 / tau) @ X ||
// nn[i]    = argmin_j ||Y_i - X_j||^2
//
// One-pass flash-softmax over j; y^2 folded out of the logits.
// S^T = Xtile @ Y^T so each query's softmax state lives in lane pair (q,q+16):
// every reduction is a single v_permlanex16. Double-buffered LDS tiles filled
// with global_load_async_to_lds_b128, software-pipelined ds_load_b128 A-frags.
// ---------------------------------------------------------------------------

typedef __attribute__((ext_vector_type(16))) __bf16 v16bf;
typedef __attribute__((ext_vector_type(8)))  __bf16 v8bf;
typedef __attribute__((ext_vector_type(8)))  float  v8f;
typedef int v4i_ __attribute__((vector_size(16)));   // matches builtin proto

#define NPT   16384
#define DIM   256
#define NTILE (NPT / 32)
#define LOG2E 1.4426950408889634f
#define TAU_F 0.07f
#define C1F   (2.0f * LOG2E / TAU_F)   // logit = g*C1F - x2scaled
#define XSCL  (LOG2E / TAU_F)          // x2scaled = ||x||^2 * XSCL

// workspace layout (needs ~16.1 MB)
#define WS_XB 0u                               // bf16 X row-major   [16384][256]
#define WS_XT (2u * NPT * DIM)                 // bf16 X feat-major  [256][16384]
#define WS_X2 (4u * NPT * DIM)                 // f32  x2scaled      [16384]

#if __has_builtin(__builtin_amdgcn_global_load_async_to_lds_b128)
#define HAS_ASYNC_LDS 1
#else
#define HAS_ASYNC_LDS 0
#endif

static __device__ __forceinline__ float fast_exp2(float x) {
#if __has_builtin(__builtin_amdgcn_exp2f)
  return __builtin_amdgcn_exp2f(x);
#else
  return exp2f(x);
#endif
}

// swap lane halves (== __shfl_xor(x,16) on wave32) without touching LDS
static __device__ __forceinline__ int swap16_i(int x) {
#if __has_builtin(__builtin_amdgcn_permlanex16)
  return __builtin_amdgcn_permlanex16(x, x, 0x76543210, 0xFEDCBA98, false, false);
#else
  return __shfl_xor(x, 16);
#endif
}
static __device__ __forceinline__ float swap16_f(float x) {
  return __int_as_float(swap16_i(__float_as_int(x)));
}

static __device__ __forceinline__ v8f wmma_bb(v16bf a, v16bf b, v8f c) {
  return __builtin_amdgcn_wmma_f32_16x16x32_bf16(false, a, false, b,
                                                 (short)0, c, false, false);
}
// identical opcode as previous WMMA, B operand unchanged -> reuse_b hint legal
static __device__ __forceinline__ v8f wmma_bb_rb(v16bf a, v16bf b, v8f c) {
  return __builtin_amdgcn_wmma_f32_16x16x32_bf16(false, a, false, b,
                                                 (short)0, c, false, true);
}

static __device__ __forceinline__ v8f zero8() {
  v8f z;
#pragma unroll
  for (int i = 0; i < 8; ++i) z[i] = 0.0f;
  return z;
}

// A-operand fragment: per-lane two contiguous 16B runs (lo at p, hi at p+32B)
static __device__ __forceinline__ v16bf load_a16(const __bf16* p) {
  v8bf lo = *(const v8bf*)(p);
  v8bf hi = *(const v8bf*)(p + 16);
  v16bf r;
#pragma unroll
  for (int i = 0; i < 8; ++i) { r[i] = lo[i]; r[8 + i] = hi[i]; }
  return r;
}

#if HAS_ASYNC_LDS
// Cast via integers (inttoptr) so clang accepts the AS-qualified pointers.
// AS1 == flat address for global memory; AS3 == low 32 bits of a flat LDS
// pointer (ISA 10.2).
static __device__ __forceinline__ void async_cp128(const void* g, void* l) {
  __builtin_amdgcn_global_load_async_to_lds_b128(
      (__attribute__((address_space(1))) v4i_*)(unsigned long long)g,
      (__attribute__((address_space(3))) v4i_*)(unsigned int)(unsigned long long)l,
      0, 0);
}
static __device__ __forceinline__ void wait_asynccnt8() {
#if __has_builtin(__builtin_amdgcn_s_wait_asynccnt)
  __builtin_amdgcn_s_wait_asynccnt(8);
#else
  asm volatile("s_wait_asynccnt 0x8" ::: "memory");
#endif
}
#endif

// stage one 32-column X tile (row-major + feat-major halves) into LDS
static __device__ __forceinline__ void stage_tile(const __bf16* __restrict__ Xb,
                                                  const __bf16* __restrict__ XT,
                                                  __bf16* sxb, __bf16* sxt,
                                                  int j0, int t) {
  const char* g1 = (const char*)(Xb + (size_t)(j0 + (t >> 3)) * DIM) + (t & 7) * 64;
  char*       l1 = (char*)(sxb + (t >> 3) * 264) + (t & 7) * 64;
  const char* g2 = (const char*)(XT + (size_t)t * NPT + j0);
  char*       l2 = (char*)(sxt + t * 40);
#if HAS_ASYNC_LDS
#pragma unroll
  for (int i = 0; i < 4; ++i) async_cp128(g1 + 16 * i, l1 + 16 * i);
#pragma unroll
  for (int i = 0; i < 4; ++i) async_cp128(g2 + 16 * i, l2 + 16 * i);
#else
#pragma unroll
  for (int i = 0; i < 4; ++i) ((uint4*)l1)[i] = ((const uint4*)g1)[i];
#pragma unroll
  for (int i = 0; i < 4; ++i) ((uint4*)l2)[i] = ((const uint4*)g2)[i];
#endif
}

// ---------------------------------------------------------------------------
// Pass 0a: X -> bf16 row-major copy, and x2scaled. One wave per row.
// ---------------------------------------------------------------------------
__global__ __launch_bounds__(256)
void prep_rows_kernel(const float* __restrict__ X, __bf16* __restrict__ Xb,
                      float* __restrict__ x2s) {
  const int lane = threadIdx.x & 31;
  const int row  = blockIdx.x * 8 + (threadIdx.x >> 5);
  const float4* src = (const float4*)(X + (size_t)row * DIM);
  float4 a = src[lane * 2 + 0];
  float4 b = src[lane * 2 + 1];
  v8bf v;
  v[0] = (__bf16)a.x; v[1] = (__bf16)a.y; v[2] = (__bf16)a.z; v[3] = (__bf16)a.w;
  v[4] = (__bf16)b.x; v[5] = (__bf16)b.y; v[6] = (__bf16)b.z; v[7] = (__bf16)b.w;
  *(v8bf*)(Xb + (size_t)row * DIM + lane * 8) = v;
  float ss = a.x*a.x + a.y*a.y + a.z*a.z + a.w*a.w
           + b.x*b.x + b.y*b.y + b.z*b.z + b.w*b.w;
#pragma unroll
  for (int off = 16; off >= 1; off >>= 1) ss += __shfl_xor(ss, off);
  if (lane == 0) x2s[row] = ss * XSCL;
}

// ---------------------------------------------------------------------------
// Pass 0b: X -> bf16 feature-major copy XT[f][j] via LDS tile transpose.
// ---------------------------------------------------------------------------
__global__ __launch_bounds__(256)
void prep_xt_kernel(const float* __restrict__ X, __bf16* __restrict__ XT) {
  __shared__ float tile[64][65];
  const int jb = blockIdx.x * 64;
  const int fb = blockIdx.y * 64;
  const int tx = threadIdx.x & 63;
  const int ty = threadIdx.x >> 6;   // 0..3
#pragma unroll
  for (int i = 0; i < 64; i += 4)
    tile[ty + i][tx] = X[(size_t)(jb + ty + i) * DIM + fb + tx];
  __syncthreads();
#pragma unroll
  for (int i = 0; i < 64; i += 4)
    XT[(size_t)(fb + ty + i) * NPT + jb + tx] = (__bf16)tile[tx][ty + i];
}

// ---------------------------------------------------------------------------
// Main fused kernel: 8 waves/block, each wave owns 16 query rows.
// ---------------------------------------------------------------------------
__global__ __launch_bounds__(256)
void flash_nn_kernel(const __bf16* __restrict__ Xb, const __bf16* __restrict__ XT,
                     const float* __restrict__ x2s, const float* __restrict__ Y,
                     float* __restrict__ outDist, int* __restrict__ outNN) {
  // double-buffered, padded LDS tiles (rows 528B / 80B -> conflict-free b128)
  __shared__ __bf16 sXb[2][32 * 264];    // X tile  row-major [32 j][256 f]
  __shared__ __bf16 sXT[2][256 * 40];    // X tile feat-major [256 f][32 j]

  const int t    = threadIdx.x;
  const int lane = t & 31;
  const int q    = lane & 15;            // query column within tile
  const int h    = lane >> 4;            // lane half
  const int row0 = (blockIdx.x * 8 + (t >> 5)) * 16;

  // Resident B operand: bf16 of this wave's 16 Y rows (K = 256 in 8 blocks)
  v16bf yb[8];
  {
    const float* yrow = Y + (size_t)(row0 + q) * DIM;
#pragma unroll
    for (int kb = 0; kb < 8; ++kb) {
      const float4* p = (const float4*)(yrow + kb * 32 + h * 8);
      float4 f0 = p[0], f1 = p[1], f2 = p[4], f3 = p[5];
      v16bf v;
      v[0]=(__bf16)f0.x; v[1]=(__bf16)f0.y; v[2]=(__bf16)f0.z; v[3]=(__bf16)f0.w;
      v[4]=(__bf16)f1.x; v[5]=(__bf16)f1.y; v[6]=(__bf16)f1.z; v[7]=(__bf16)f1.w;
      v[8]=(__bf16)f2.x; v[9]=(__bf16)f2.y; v[10]=(__bf16)f2.z; v[11]=(__bf16)f2.w;
      v[12]=(__bf16)f3.x; v[13]=(__bf16)f3.y; v[14]=(__bf16)f3.z; v[15]=(__bf16)f3.w;
      yb[kb] = v;
    }
  }

  v8f acc[16];                           // acc^T: [256 f][16 q] in 16 D-tiles
#pragma unroll
  for (int i = 0; i < 16; ++i) acc[i] = zero8();

  float mrun = -__builtin_inff();        // running max (pair-uniform)
  float srun = 0.0f;                     // per-lane partial sum
  float bestv = -__builtin_inff();       // per-lane argmax candidate
  int   besti = 0;

  stage_tile(Xb, XT, sXb[0], sXT[0], 0, t);   // prologue prefetch

  for (int jt = 0; jt < NTILE; ++jt) {
    const int cur = jt & 1;
    const int j0  = jt * 32;

    // prefetch next tile into the other buffer (wraps on last iter; harmless)
    stage_tile(Xb, XT, sXb[cur ^ 1], sXT[cur ^ 1],
               ((jt + 1) & (NTILE - 1)) * 32, t);
#if HAS_ASYNC_LDS
    wait_asynccnt8();                    // retire current tile's async group
#endif
    __syncthreads();

    const __bf16* xb = sXb[cur];
    const __bf16* xt = sXT[cur];

    // ---- S^T = Xtile(32x256) @ Y^T(256x16), software-pipelined A frags ----
    v8f S0 = zero8(), S1 = zero8();
    {
      v16bf a = load_a16(xb + q * 264 + h * 8);   // frag k=0
#pragma unroll
      for (int k = 0; k < 16; ++k) {
        v16bf an = a;
        if (k < 15) {
          const int kn = k + 1;
          an = load_a16(xb + (((kn >> 3) << 4) + q) * 264 + (kn & 7) * 32 + h * 8);
        }
        if (k < 8) S0 = wmma_bb(a, yb[k], S0);
        else       S1 = wmma_bb(a, yb[k - 8], S1);
        a = an;
      }
    }

    // ---- logits + online softmax state (pair-local) ----------------------
    const float* xs = x2s + j0;          // uniform -> s_load
    float l0[8], l1v[8];
    float cmax = -__builtin_inff();
#pragma unroll
    for (int g = 0; g < 8; ++g) {
      float xv0 = h ? xs[g + 8]  : xs[g];
      float xv1 = h ? xs[g + 24] : xs[g + 16];
      l0[g]  = S0[g] * C1F - xv0;        // j = j0      + g + 8h
      l1v[g] = S1[g] * C1F - xv1;        // j = j0 + 16 + g + 8h
      cmax = fmaxf(cmax, fmaxf(l0[g], l1v[g]));
    }
    cmax = fmaxf(cmax, swap16_f(cmax));
    const float mold = mrun;
    const float mnew = fmaxf(mold, cmax);
    mrun = mnew;

    float p0[8], p1[8], psum = 0.0f;
#pragma unroll
    for (int g = 0; g < 8; ++g) {
      p0[g] = fast_exp2(l0[g]  - mnew);
      p1[g] = fast_exp2(l1v[g] - mnew);
      psum += p0[g] + p1[g];
      const int jg0 = j0 + g + 8 * h;
      const int jg1 = j0 + 16 + g + 8 * h;
      if (l0[g]  > bestv) { bestv = l0[g];  besti = jg0; }
      if (l1v[g] > bestv) { bestv = l1v[g]; besti = jg1; }
    }

    if (__any(mnew > mold)) {            // rescale only when a row max moved
      const float scale = fast_exp2(mold - mnew);
      srun = srun * scale + psum;
#pragma unroll
      for (int i = 0; i < 16; ++i)
#pragma unroll
        for (int g = 0; g < 8; ++g) acc[i][g] *= scale;
    } else {
      srun += psum;
    }

    // ---- pack P^T into a B operand (32 j x 16 q) from lane-local p0/p1 -------
    v16bf bP;
#pragma unroll
    for (int g = 0; g < 8; ++g) {
      const float x0 = swap16_f(p0[g]);
      const float x1 = swap16_f(p1[g]);
      bP[g]     = (__bf16)(h ? p0[g] : p0[g]);   // K = e    (h=0) / 8+e  (h=1)
      bP[8 + g] = (__bf16)(h ? p1[g] : p1[g]);   // K = 16+e (h=0) / 24+e (h=1)
    }

    // ---- acc^T += Xtile^T(256x32) @ P^T(32x16); B reused -> reuse_b hint -
    {
      v16bf a = load_a16(xt + q * 40 + h * 8);   // frag i=0
      {
        v16bf an = load_a16(xt + (16 + q) * 40 + h * 8);
        acc[0] = wmma_bb(a, bP, acc[0]);
        a = an;
      }
#pragma unroll
      for (int i = 1; i < 16; ++i) {
        v16bf an = a;
        if (i < 15) an = load_a16(xt + (16 * (i + 1) + q) * 40 + h * 8);
        acc[i] = wmma_bb_rb(a, bP, acc[i]);
        a = an;
      }
    }

    __syncthreads();                     // all reads done before overwrite
  }

  // ---- finalize: pair-combine stats, distance, outputs -------------------
  const float stot = srun + swap16_f(srun);
  {
    const float ov = swap16_f(bestv);
    const int   oi = swap16_i(besti);
    if (ov > bestv || (ov == bestv && oi < besti)) { bestv = ov; besti = oi; }
  }
  const float inv = 1.0f / stot;
  float d2 = 0.0f;
  const float* yrow = Y + (size_t)(row0 + q) * DIM;
#pragma unroll
  for (int i = 0; i < 16; ++i) {
    const float4* p = (const float4*)(yrow + 16 * i + 8 * h);
    float4 f0 = p[0], f1 = p[1];
    const float yv[8] = {f0.x, f0.y, f0.z, f0.w, f1.x, f1.y, f1.z, f1.w};
#pragma unroll
    for (int g = 0; g < 8; ++g) {
      const float d = yv[g] - acc[i][g] * inv;   // feat = 16i + g + 8h
      d2 += d * d;
    }
  }
  d2 += swap16_f(d2);
  if (h == 0) {
    outDist[row0 + q] = sqrtf(d2);
    outNN[row0 + q]   = besti;
  }
}

// ---------------------------------------------------------------------------
extern "C" void kernel_launch(void* const* d_in, const int* in_sizes, int n_in,
                              void* d_out, int out_size, void* d_ws, size_t ws_size,
                              hipStream_t stream) {
  (void)in_sizes; (void)n_in; (void)out_size; (void)ws_size; // needs ~16.1MB ws
  const float* X = (const float*)d_in[0];
  const float* Y = (const float*)d_in[1];
  char* ws = (char*)d_ws;
  __bf16* Xb  = (__bf16*)(ws + WS_XB);
  __bf16* XT  = (__bf16*)(ws + WS_XT);
  float*  x2s = (float*)(ws + WS_X2);

  prep_rows_kernel<<<NPT / 8, 256, 0, stream>>>(X, Xb, x2s);
  prep_xt_kernel<<<dim3(NPT / 64, DIM / 64), 256, 0, stream>>>(X, XT);

  float* outDist = (float*)d_out;                 // dists: 16384 f32
  int*   outNN   = (int*)d_out + NPT;             // nn:    16384 i32 (flat after)
  flash_nn_kernel<<<NPT / 128, 256, 0, stream>>>(Xb, XT, x2s, Y, outDist, outNN);
}
